// Het_Agg_49323404427473
// MI455X (gfx1250) — matrix operand order repaired and checked by reading, not compiled, not run on hardware
//
#include <hip/hip_runtime.h>
#include <hip/hip_bf16.h>

// ---------------------------------------------------------------------------
// CDNA5 (gfx1250) heterogeneous GNN aggregation layer.
//  - GEMMs via v_wmma_f32_16x16x32_bf16, fp32 emulated with bf16 hi/lo split
//  - LDS tiles stored in *fragment order*: each lane's 16-element WMMA operand
//    is 32 contiguous bytes -> ds_load_b128 x2 instead of 16x ds_load_u16
//  - edge scatter-add via L2 f32 atomics (aggr ~102MB, fits 192MB L2)
//  - wave32 shuffle reductions for attention dots and row norms
// ---------------------------------------------------------------------------

typedef __attribute__((ext_vector_type(16))) __bf16 v16bf;
typedef __attribute__((ext_vector_type(2)))  __bf16 v2bf;
typedef __attribute__((ext_vector_type(8)))  float  v8f;

#define MAXCHUNKS 8   // supports K up to 256

// Y[N,128] = relu(X[N,K] @ W[K,128] + bias).
// Block = 256 threads = 8 waves; wave wv owns output columns [16wv,16wv+16).
//
// A-fragment layout (16x32 bf16, ISA 7.12.2): lane<16 -> M=lane, K={0..7,16..23};
// lane>=16 -> M=lane-16, K={8..15,24..31}. Inverse used at staging time:
//   dlane = r + 16*((kl>>3)&1),  j = (kl&7) + 8*(kl>>4)
// B-fragment layout (32x16): lane<16 -> N=lane, K={0..7,16..23}; lane>=16 -> K={8..15,24..31}:
//   dlane = (c&15) + 16*((kl>>3)&1), j = (kl&7) + 8*(kl>>4)
// Pairs (kl, kl+1) with kl even always map to (j, j+1) -> packed b32 stores.
__global__ __launch_bounds__(256)
void gemm_bias_relu_kernel(const float* __restrict__ X,
                           const float* __restrict__ W,
                           const float* __restrict__ bias,
                           float* __restrict__ Y,
                           int Nrows, int K) {
  __shared__ __align__(32) __bf16 sAhi[MAXCHUNKS * 32 * 16];
  __shared__ __align__(32) __bf16 sAlo[MAXCHUNKS * 32 * 16];
  __shared__ __align__(32) __bf16 sBhi[8 * 32 * 16];   // 8 column tiles
  __shared__ __align__(32) __bf16 sBlo[8 * 32 * 16];

  const int tid   = threadIdx.x;
  const int lane  = tid & 31;
  const int wv    = tid >> 5;
  const int row0  = blockIdx.x * 16;
  const int kpair = K >> 1;

  // Stage X tile (16 x K) in A-fragment order, split bf16, two k per thread.
  for (int i = tid; i < 16 * kpair; i += 256) {
    const int r = i / kpair;
    const int k = (i - r * kpair) * 2;
    int gr = row0 + r; if (gr >= Nrows) gr = Nrows - 1;   // clamp tail reads
    const float2 xv = *(const float2*)(X + (size_t)gr * K + k);
    const int chunk = k >> 5;
    const int kl    = k & 31;
    const int dlane = r + (((kl >> 3) & 1) << 4);
    const int j     = (kl & 7) + ((kl >> 4) << 3);
    const int idx   = (chunk * 32 + dlane) * 16 + j;      // j even -> 4B aligned
    const __bf16 h0 = (__bf16)xv.x, h1 = (__bf16)xv.y;
    v2bf ph; ph[0] = h0; ph[1] = h1;
    v2bf pl; pl[0] = (__bf16)(xv.x - (float)h0);
             pl[1] = (__bf16)(xv.y - (float)h1);
    *(v2bf*)&sAhi[idx] = ph;
    *(v2bf*)&sAlo[idx] = pl;
  }

  v8f acc0 = {};   // hi*hi chain
  v8f acc1 = {};   // cross-term chain (hi*lo + lo*hi)
  const int afrag = lane * 16;
  const int bfrag = (wv * 32 + lane) * 16;

  for (int kk = 0; kk < K; kk += 32) {
    __syncthreads();
    // Stage W chunk (32 x 128) in B-fragment order, split bf16.
    for (int i = tid; i < (32 * 128) / 2; i += 256) {
      const int c  = i & 127;
      const int kl = (i >> 7) << 1;
      const float w0 = W[(size_t)(kk + kl) * 128 + c];
      const float w1 = W[(size_t)(kk + kl + 1) * 128 + c];
      const int tile  = c >> 4;
      const int dlane = (c & 15) + (((kl >> 3) & 1) << 4);
      const int j     = (kl & 7) + ((kl >> 4) << 3);       // even
      const int idx   = (tile * 32 + dlane) * 16 + j;
      const __bf16 h0 = (__bf16)w0, h1 = (__bf16)w1;
      v2bf ph; ph[0] = h0; ph[1] = h1;
      v2bf pl; pl[0] = (__bf16)(w0 - (float)h0);
               pl[1] = (__bf16)(w1 - (float)h1);
      *(v2bf*)&sBhi[idx] = ph;
      *(v2bf*)&sBlo[idx] = pl;
    }
    __syncthreads();

    // Contiguous 32B fragment loads -> 2x ds_load_b128 each.
    const int cbase = (kk >> 5) * 32 * 16;
    const v16bf ahi = *(const v16bf*)&sAhi[cbase + afrag];
    const v16bf alo = *(const v16bf*)&sAlo[cbase + afrag];
    const v16bf bhi = *(const v16bf*)&sBhi[bfrag];
    const v16bf blo = *(const v16bf*)&sBlo[bfrag];

    // x*w ~= hi*hi + hi*lo + lo*hi  (drop lo*lo; ~2^-16 relative error)
    acc0 = __builtin_amdgcn_wmma_f32_16x16x32_bf16(false, ahi, false, bhi,
                                                   (short)0, acc0, false, false);
    acc1 = __builtin_amdgcn_wmma_f32_16x16x32_bf16(false, ahi, false, blo,
                                                   (short)0, acc1, false, false);
    acc1 = __builtin_amdgcn_wmma_f32_16x16x32_bf16(false, alo, false, bhi,
                                                   (short)0, acc1, false, false);
  }

  // C/D layout: VGPR v -> M = (lane>=16)*8 + v, N = lane&15.
  const int half = lane >> 4;
  const int col  = wv * 16 + (lane & 15);
  const float b  = bias[col];
  #pragma unroll
  for (int v = 0; v < 8; ++v) {
    const int gr = row0 + half * 8 + v;
    if (gr < Nrows) {
      const float o = acc0[v] + acc1[v] + b;
      Y[(size_t)gr * 128 + col] = o > 0.f ? o : 0.f;
    }
  }
}

// One wave per edge: lane handles a float4 chunk of the 128-wide row.
// aggr[src] += H[tgt]; deg[src] += 1.  Atomics land in L2.
__global__ __launch_bounds__(256)
void edge_agg_kernel(const int* __restrict__ src,
                     const int* __restrict__ tgt,
                     const float* __restrict__ H,
                     float* __restrict__ aggr,
                     float* __restrict__ deg,
                     int E) {
  const int lane   = threadIdx.x & 31;
  const int wave   = (int)((blockIdx.x * blockDim.x + threadIdx.x) >> 5);
  const int nwaves = (int)((gridDim.x * blockDim.x) >> 5);
  for (int e = wave; e < E; e += nwaves) {
    const int s = src[e];
    const int t = tgt[e];
    const int en = e + nwaves;
    if (en < E) {  // prefetch next iteration's gather row (global_prefetch_b8)
      const int tn = tgt[en];
      __builtin_prefetch(H + (size_t)tn * 128 + lane * 4, 0, 1);
    }
    const float4 v = *reinterpret_cast<const float4*>(H + (size_t)t * 128 + lane * 4);
    float* dst = aggr + (size_t)s * 128 + lane * 4;
    atomicAdd(dst + 0, v.x);
    atomicAdd(dst + 1, v.y);
    atomicAdd(dst + 2, v.z);
    atomicAdd(dst + 3, v.w);
    if (lane == 0) atomicAdd(deg + s, 1.0f);
  }
}

// One wave per node row: degree-normalize, attention softmax over 4 relations,
// write concat = [x_node, combined] for the final GEMM.
__global__ __launch_bounds__(256)
void combine_kernel(const float* __restrict__ aggr,   // [4][N][128]
                    const float* __restrict__ deg,    // [4][N]
                    const float* __restrict__ x_node, // [N][128]
                    const float* __restrict__ u,      // [256]
                    float* __restrict__ concat,       // [N][256]
                    int N) {
  const int lane = threadIdx.x & 31;
  const int row  = (int)((blockIdx.x * blockDim.x + threadIdx.x) >> 5);
  if (row >= N) return;
  const int c4 = lane * 4;

  const float4 xn  = *(const float4*)(x_node + (size_t)row * 128 + c4);
  const float4 ulo = *(const float4*)(u + c4);
  const float4 uhi = *(const float4*)(u + 128 + c4);
  float px = xn.x * uhi.x + xn.y * uhi.y + xn.z * uhi.z + xn.w * uhi.w;

  float4 a[4];
  float  pr[4];
  #pragma unroll
  for (int r = 0; r < 4; ++r) {
    float d = deg[(size_t)r * N + row];
    d = d < 1.f ? 1.f : d;
    const float inv = 1.f / d;
    float4 t = *(const float4*)(aggr + ((size_t)r * N + row) * 128 + c4);
    t.x *= inv; t.y *= inv; t.z *= inv; t.w *= inv;
    a[r]  = t;
    pr[r] = t.x * ulo.x + t.y * ulo.y + t.z * ulo.z + t.w * ulo.w;
  }

  #pragma unroll
  for (int off = 16; off > 0; off >>= 1) {
    px += __shfl_xor(px, off, 32);
    #pragma unroll
    for (int r = 0; r < 4; ++r) pr[r] += __shfl_xor(pr[r], off, 32);
  }

  float sc[4], denom = 0.f;
  #pragma unroll
  for (int r = 0; r < 4; ++r) {
    float z = pr[r] + px;
    z = z > 0.f ? z : 0.01f * z;          // leaky_relu
    sc[r] = __expf(z);
    denom += sc[r];
  }
  const float invd = 1.f / denom;
  float4 comb = {0.f, 0.f, 0.f, 0.f};
  #pragma unroll
  for (int r = 0; r < 4; ++r) {
    const float w = sc[r] * invd;
    comb.x += w * a[r].x; comb.y += w * a[r].y;
    comb.z += w * a[r].z; comb.w += w * a[r].w;
  }

  float* out = concat + (size_t)row * 256;
  *(float4*)(out + c4)       = xn;
  *(float4*)(out + 128 + c4) = comb;
}

// Row-wise L2 normalize in place: y /= max(||y||, 1e-12).
__global__ __launch_bounds__(256)
void normalize_kernel(float* __restrict__ Y, int N) {
  const int lane = threadIdx.x & 31;
  const int row  = (int)((blockIdx.x * blockDim.x + threadIdx.x) >> 5);
  if (row >= N) return;
  float4 v = *(float4*)(Y + (size_t)row * 128 + lane * 4);
  float ss = v.x * v.x + v.y * v.y + v.z * v.z + v.w * v.w;
  #pragma unroll
  for (int off = 16; off > 0; off >>= 1) ss += __shfl_xor(ss, off, 32);
  float n = sqrtf(ss);
  n = n < 1e-12f ? 1e-12f : n;
  const float inv = 1.f / n;
  v.x *= inv; v.y *= inv; v.z *= inv; v.w *= inv;
  *(float4*)(Y + (size_t)row * 128 + lane * 4) = v;
}

extern "C" void kernel_launch(void* const* d_in, const int* in_sizes, int n_in,
                              void* d_out, int out_size, void* d_ws, size_t ws_size,
                              hipStream_t stream) {
  // setup_inputs() dict order:
  // 0:x_g 1:edges_g 2:x_d 3:edges_d 4:x_c 5:edges_c 6:x_s 7:edges_s
  // 8:x_node 9:num_node 10..17:(W_r,b_r)*4 18:u 19:W_lin 20:b_lin
  const float* x[4]     = {(const float*)d_in[0], (const float*)d_in[2],
                           (const float*)d_in[4], (const float*)d_in[6]};
  const int*   edges[4] = {(const int*)d_in[1], (const int*)d_in[3],
                           (const int*)d_in[5], (const int*)d_in[7]};
  const float* x_node   = (const float*)d_in[8];
  const float* Wr[4]    = {(const float*)d_in[10], (const float*)d_in[12],
                           (const float*)d_in[14], (const float*)d_in[16]};
  const float* br[4]    = {(const float*)d_in[11], (const float*)d_in[13],
                           (const float*)d_in[15], (const float*)d_in[17]};
  const float* u        = (const float*)d_in[18];
  const float* W_lin    = (const float*)d_in[19];
  const float* b_lin    = (const float*)d_in[20];

  const int N = in_sizes[0] / 128;   // 50000
  const int E = in_sizes[1] / 2;     // 1600000

  // Workspace layout (floats): H | aggr[4] | deg[4] | concat
  float* ws     = (float*)d_ws;
  float* H      = ws;                                  // N*128 (reused per relation)
  float* aggr   = H + (size_t)N * 128;                 // 4*N*128
  float* deg    = aggr + (size_t)4 * N * 128;          // 4*N
  float* concat = deg + (size_t)4 * N;                 // N*256

  // aggr+deg are contiguous: zero both each call (graph-capture safe).
  hipMemsetAsync(aggr, 0, ((size_t)4 * N * 128 + (size_t)4 * N) * sizeof(float),
                 stream);

  const int gemmBlocks = (N + 15) / 16;
  for (int r = 0; r < 4; ++r) {
    gemm_bias_relu_kernel<<<gemmBlocks, 256, 0, stream>>>(
        x[r], Wr[r], br[r], H, N, 128);
    edge_agg_kernel<<<4096, 256, 0, stream>>>(
        edges[r], edges[r] + E, H,
        aggr + (size_t)r * N * 128, deg + (size_t)r * N, E);
  }

  const int rowBlocks = (N + 7) / 8;
  combine_kernel<<<rowBlocks, 256, 0, stream>>>(aggr, deg, x_node, u, concat, N);
  gemm_bias_relu_kernel<<<gemmBlocks, 256, 0, stream>>>(
      concat, W_lin, b_lin, (float*)d_out, N, 256);
  normalize_kernel<<<rowBlocks, 256, 0, stream>>>((float*)d_out, N);
}
